// _PatchPDE_43181601194897
// MI455X (gfx1250) — matrix operand order repaired and checked by reading, not compiled, not run on hardware
//
#include <hip/hip_runtime.h>
#include <math.h>

// ---------------------------------------------------------------------------
// CDNA5 WMMA (wave32): D(16x16 f32) = A(16x32 bf16) x B(32x16 bf16) + C
// ---------------------------------------------------------------------------
typedef __attribute__((ext_vector_type(16))) __bf16        v16bf;
typedef __attribute__((ext_vector_type(2)))  __bf16        v2bf;
typedef __attribute__((ext_vector_type(8)))  float         v8f;
typedef __attribute__((ext_vector_type(4)))  unsigned int  v4u;

union FragCvt { v4u q[2]; v16bf v; };

__device__ __forceinline__ unsigned int bf16_rne(unsigned int u) {
  return u + 0x7FFFu + ((u >> 16) & 1u);    // round-to-nearest-even in [31:16]
}
__device__ __forceinline__ unsigned short f2bf(float f) {
  return (unsigned short)(bf16_rne(__builtin_bit_cast(unsigned int, f)) >> 16);
}
__device__ __forceinline__ unsigned int f2bf2(float a, float b) {
#if __has_builtin(__builtin_amdgcn_cvt_pk_bf16_f32)
  v2bf p = __builtin_amdgcn_cvt_pk_bf16_f32(a, b);
  return __builtin_bit_cast(unsigned int, p);
#else
  unsigned int ua = bf16_rne(__builtin_bit_cast(unsigned int, a));
  unsigned int ub = bf16_rne(__builtin_bit_cast(unsigned int, b));
  return (ua >> 16) | (ub & 0xFFFF0000u);   // byte-merge (v_perm class)
#endif
}
__device__ __forceinline__ v4u pack8_bf16(v8f s) {
  v4u r;
  r.x = f2bf2(s[0], s[1]);  r.y = f2bf2(s[2], s[3]);
  r.z = f2bf2(s[4], s[5]);  r.w = f2bf2(s[6], s[7]);
  return r;
}

__device__ __forceinline__ v8f wmma_bf16(v16bf a, v16bf b, v8f c) {
  return __builtin_amdgcn_wmma_f32_16x16x32_bf16(false, a, false, b, (short)0, c,
                                                 false, false);
}

// A-operand fragment, matrix row-major in LDS (ld mult of 8, col0 mult of 32).
__device__ __forceinline__ v16bf ldsA_frag(const unsigned short* p, int row0,
                                           int col0, int ld, int lane) {
  const int m = lane & 15, half = lane >> 4;
  const unsigned short* base = p + (row0 + m) * ld + col0 + 8 * half;
  FragCvt u;
  u.q[0] = *(const v4u*)(base);        // ds_load_b128
  u.q[1] = *(const v4u*)(base + 16);
  return u.v;
}
// B-operand fragment from TRANSPOSED storage: pT[n][k] = B[k][n].
__device__ __forceinline__ v16bf ldsBt_frag(const unsigned short* pT, int n0,
                                            int k0, int ld, int lane) {
  const int n = lane & 15, half = lane >> 4;
  const unsigned short* base = pT + (n0 + n) * ld + k0 + 8 * half;
  FragCvt u;
  u.q[0] = *(const v4u*)(base);
  u.q[1] = *(const v4u*)(base + 16);
  return u.v;
}

__device__ __forceinline__ float softplus_clamped(float x) {
  float sp = (x > 20.f) ? x : log1pf(expf(x));
  return fminf(sp, 2.0f);
}

#define BN_TOTAL 4096
#define P 64
#define D 128
#define KK 9
#define LD_X 136   // xn bf16 stride
#define LD_A 72    // A bf16 stride (symmetric; serves A- and B-operand roles)
#define LD_W 136   // W' stride:  WT[d'][d] = theta[d'][d] - alpha*(d'==d)
#define LD_T 72    // tT[d][p] = t[p][d]   (p contiguous)
#define LD_R 136   // R [p][d] = r[p][d]   (d contiguous)
#define LD_Y 132   // y f32 stride
#define LD_C 66    // corr f32 stride

// ---------------------------------------------------------------------------
// Kernel 0: WT[d'][d] = bf16( theta[d'][d] - alpha*(d'==d) )  == theta row-major
// ---------------------------------------------------------------------------
__global__ void prep_wp_kernel(const float* __restrict__ theta,
                               const float* __restrict__ alpha_raw,
                               unsigned short* __restrict__ wpT) {
  float alpha = softplus_clamped(alpha_raw[0]);
  int i = blockIdx.x * blockDim.x + threadIdx.x;
  if (i < D * D) {
    int n = i >> 7, k = i & 127;
    wpT[i] = f2bf(theta[i] - ((n == k) ? alpha : 0.f));
  }
}

// ---------------------------------------------------------------------------
// Kernel 1: sparse normalized graph A per (b,n); corr GEMM via WMMA bf16
// ---------------------------------------------------------------------------
__global__ __launch_bounds__(256) void build_A_kernel(
    const float* __restrict__ pt, unsigned short* __restrict__ Aout) {
  __shared__ float xf[P * LD_Y];
  __shared__ unsigned short xnbf[P * LD_X];
  __shared__ float corr[P * LD_C];
  __shared__ float rowmean[P], rowisd[P], dinv[P];
  __shared__ unsigned long long maskbits[P];

  const int tid = threadIdx.x, lane = tid & 31, wave = tid >> 5;
  const long long bn = blockIdx.x;
  const float* src = pt + bn * (long long)(P * D);

  for (int i = tid; i < P * D / 4; i += 256)
    *(float4*)&xf[(i >> 5) * LD_Y + (i & 31) * 4] = ((const float4*)src)[i];
  __syncthreads();

  { // row mean / unbiased std
    int row = tid >> 2, part = tid & 3;
    const float* rp = &xf[row * LD_Y];
    float s = 0.f;
    for (int c = part * 32; c < part * 32 + 32; ++c) s += rp[c];
    s += __shfl_xor(s, 1);  s += __shfl_xor(s, 2);
    float mean = s * (1.0f / D);
    float v = 0.f;
    for (int c = part * 32; c < part * 32 + 32; ++c) {
      float d = rp[c] - mean; v += d * d;
    }
    v += __shfl_xor(v, 1);  v += __shfl_xor(v, 2);
    if (part == 0) {
      rowmean[row] = mean;
      rowisd[row] = 1.0f / (sqrtf(v / (D - 1)) + 1e-6f);
    }
  }
  __syncthreads();

  for (int i = tid; i < P * D / 2; i += 256) {
    int r = i >> 6, c2 = (i & 63) * 2;
    float mu = rowmean[r], is = rowisd[r];
    *(unsigned int*)&xnbf[r * LD_X + c2] =
        f2bf2((xf[r * LD_Y + c2] - mu) * is, (xf[r * LD_Y + c2 + 1] - mu) * is);
  }
  __syncthreads();

  const float cscale = 1.0f / ((float)(D - 1) + 1e-6f);
#pragma unroll
  for (int t = 0; t < 2; ++t) {
    int tile = wave * 2 + t, m = tile >> 2, n = tile & 3;
    v8f c = {0.f, 0.f, 0.f, 0.f, 0.f, 0.f, 0.f, 0.f};
#pragma unroll
    for (int k = 0; k < 4; ++k) {
      v16bf a = ldsA_frag(xnbf, 16 * m, 32 * k, LD_X, lane);
      v16bf b = ldsBt_frag(xnbf, 16 * n, 32 * k, LD_X, lane);  // B = xn^T
      c = wmma_bf16(a, b, c);
    }
    int half = lane >> 4, nn = lane & 15;
#pragma unroll
    for (int j = 0; j < 8; ++j)
      corr[(16 * m + j + 8 * half) * LD_C + 16 * n + nn] = c[j] * cscale;
  }
  __syncthreads();

  if (tid < P) {  // top-(k+1), first-index tie-break like lax.top_k
    const float* rowp = &corr[tid * LD_C];
    unsigned long long m = 0ull;
#pragma unroll 1
    for (int s = 0; s < KK; ++s) {
      float best = -INFINITY; int bj = 0;
      for (int j = 0; j < P; ++j)
        if (!((m >> j) & 1ull)) {
          float v = rowp[j];
          if (v > best) { best = v; bj = j; }
        }
      m |= (1ull << bj);
    }
    maskbits[tid] = m;
  }
  __syncthreads();

  for (int i = tid; i < P * P; i += 256) {  // symmetrize + I (in-place)
    int r = i >> 6, c = i & 63;
    float mi = (float)((maskbits[r] >> c) & 1ull);
    float mj = (float)((maskbits[c] >> r) & 1ull);
    float a = corr[r * LD_C + c] * 0.5f * (mi + mj);
    if (r == c) a += 1.0f;
    corr[r * LD_C + c] = a;
  }
  __syncthreads();

  if (tid < P) {
    float s = 0.f;
    for (int j = 0; j < P; ++j) s += corr[tid * LD_C + j];
    dinv[tid] = 1.0f / sqrtf(fmaxf(s, 1e-6f));
  }
  __syncthreads();

  unsigned short* dstA = Aout + bn * (long long)(P * P);
  for (int i = tid; i < P * P / 8; i += 256) {
    int row = i >> 3, c0 = (i & 7) * 8;
    float dr = dinv[row];
    const float* cr = &corr[row * LD_C + c0];
    v4u pk;
    pk.x = f2bf2(dr * cr[0] * dinv[c0 + 0], dr * cr[1] * dinv[c0 + 1]);
    pk.y = f2bf2(dr * cr[2] * dinv[c0 + 2], dr * cr[3] * dinv[c0 + 3]);
    pk.z = f2bf2(dr * cr[4] * dinv[c0 + 4], dr * cr[5] * dinv[c0 + 5]);
    pk.w = f2bf2(dr * cr[6] * dinv[c0 + 6], dr * cr[7] * dinv[c0 + 7]);
    ((v4u*)dstA)[i] = pk;
  }
}

// ---------------------------------------------------------------------------
// Kernel 2: fused RK4.  RHS(t) = alpha*t + (A t) W'.
//   mm1:  r^T = t^T * A      (A symmetric)  -> R[p][d], d-contiguous b128 store
//   mm2:  s   = R * W'                       -> lanes run along d'
// y, k_prev, RK4 accumulator, and the A-graph B-fragments all in registers.
// ---------------------------------------------------------------------------
__global__ __launch_bounds__(256) void ode_rk4_kernel(
    const float* __restrict__ pt, const unsigned short* __restrict__ Aglob,
    const unsigned short* __restrict__ WpT, const float* __restrict__ alpha_raw,
    float* __restrict__ out) {
  __shared__ float y[P * LD_Y];                 // init staging + final store
  __shared__ unsigned short Abf[P * LD_A];
  __shared__ unsigned short WT[D * LD_W];
  __shared__ unsigned short tT[D * LD_T];       // tT[d][p]
  __shared__ unsigned short R[P * LD_R];        // R[p][d]     total ~114 KB

  const int tid = threadIdx.x, lane = tid & 31, wave = tid >> 5;
  const long long bn = blockIdx.x;
  const float alpha = softplus_clamped(alpha_raw[0]);
  const float h = 0.2f / 8.0f;

  const float* src = pt + bn * (long long)(P * D);
  for (int i = tid; i < P * D / 4; i += 256)
    *(float4*)&y[(i >> 5) * LD_Y + (i & 31) * 4] = ((const float4*)src)[i];
  const unsigned short* As = Aglob + bn * (long long)(P * P);
  for (int i = tid; i < P * P / 8; i += 256)
    *(v4u*)&Abf[(i >> 3) * LD_A + (i & 7) * 8] = ((const v4u*)As)[i];
  for (int i = tid; i < D * D / 8; i += 256)
    *(v4u*)&WT[(i >> 4) * LD_W + (i & 15) * 8] = ((const v4u*)WpT)[i];
  __syncthreads();

  // initial operand tT[d][p] = bf16(y[p][d])
  for (int i = tid; i < P * D / 2; i += 256) {
    int d = i >> 5, p2 = (i & 31) * 2;
    *(unsigned int*)&tT[d * LD_T + p2] =
        f2bf2(y[p2 * LD_Y + d], y[(p2 + 1) * LD_Y + d]);
  }

  const int W3 = wave & 3, W4 = (wave >> 2) * 4;
  const int half = lane >> 4, nn = lane & 15;
  // owned element (q,j):  p = 16*W3 + 8*half + j ,  d' = 16*(W4+q) + nn
  v8f yreg[4], acc[4], kprev[4];
#pragma unroll
  for (int q = 0; q < 4; ++q)
#pragma unroll
    for (int j = 0; j < 8; ++j)
      yreg[q][j] = y[(16 * W3 + 8 * half + j) * LD_Y + 16 * (W4 + q) + nn];

  // mm1 B-operand fragments: the graph A is loop-invariant -> registers
  v16bf bA[2];
  bA[0] = ldsBt_frag(Abf, 16 * W3, 0, LD_A, lane);
  bA[1] = ldsBt_frag(Abf, 16 * W3, 32, LD_A, lane);
  __syncthreads();

#pragma unroll 1
  for (int step = 0; step < 8; ++step) {
#pragma unroll 1
    for (int st = 0; st < 4; ++st) {
      // mm1: r^T = t^T * A  -> R[p][d]  (M=d tiles W4..W4+3, N=p tile W3)
      v8f rs[4];
#pragma unroll
      for (int q = 0; q < 4; ++q) rs[q] = (v8f){0.f,0.f,0.f,0.f,0.f,0.f,0.f,0.f};
#pragma unroll
      for (int k = 0; k < 2; ++k)
#pragma unroll
        for (int q = 0; q < 4; ++q) {
          v16bf a = ldsA_frag(tT, 16 * (W4 + q), 32 * k, LD_T, lane);
          rs[q] = wmma_bf16(a, bA[k], rs[q]);
        }
#pragma unroll
      for (int q = 0; q < 4; ++q)   // one b128 store per tile
        *(v4u*)&R[(16 * W3 + nn) * LD_R + 16 * (W4 + q) + 8 * half] =
            pack8_bf16(rs[q]);
      __syncthreads();

      // mm2: s = R * W'  (M=p tile W3, N=d' tiles W4..W4+3, K-loop of 4)
      v8f ss[4];
#pragma unroll
      for (int q = 0; q < 4; ++q) ss[q] = (v8f){0.f,0.f,0.f,0.f,0.f,0.f,0.f,0.f};
#pragma unroll
      for (int k = 0; k < 4; ++k) {
        v16bf a = ldsA_frag(R, 16 * W3, 32 * k, LD_R, lane);
#pragma unroll
        for (int q = 0; q < 4; ++q) {
          v16bf b = ldsBt_frag(WT, 16 * (W4 + q), 32 * k, LD_W, lane);
          ss[q] = wmma_bf16(a, b, ss[q]);
        }
      }

      // fused RK4 update; emit next stage's operand into tT (b128 per tile)
      const float cin = (st == 3) ? h : 0.5f * h;
      const float cnext = (st == 2) ? h : 0.5f * h;
#pragma unroll
      for (int q = 0; q < 4; ++q) {
        v8f tn;
#pragma unroll
        for (int j = 0; j < 8; ++j) {
          float yv = yreg[q][j];
          float t = (st == 0) ? yv : fmaf(cin, kprev[q][j], yv);
          float kv = fmaf(alpha, t, ss[q][j]);
          if (st < 3) {
            acc[q][j] = (st == 0) ? kv : acc[q][j] + 2.0f * kv;
            kprev[q][j] = kv;
            tn[j] = fmaf(cnext, kv, yv);
          } else {
            float yn = fmaf(h / 6.0f, acc[q][j] + kv, yv);
            yreg[q][j] = yn;
            tn[j] = yn;
          }
        }
        if (st < 3 || step < 7)
          *(v4u*)&tT[(16 * (W4 + q) + nn) * LD_T + 16 * W3 + 8 * half] =
              pack8_bf16(tn);
      }
      __syncthreads();
    }
  }

  // write back owned state, then cooperative relu + vectorized global store
#pragma unroll
  for (int q = 0; q < 4; ++q)
#pragma unroll
    for (int j = 0; j < 8; ++j)
      y[(16 * W3 + 8 * half + j) * LD_Y + 16 * (W4 + q) + nn] = yreg[q][j];
  __syncthreads();

  float* dst = out + bn * (long long)(P * D);
  for (int i = tid; i < P * D / 4; i += 256) {
    float4 v = *(const float4*)&y[(i >> 5) * LD_Y + (i & 31) * 4];
    v.x = fmaxf(v.x, 0.f); v.y = fmaxf(v.y, 0.f);
    v.z = fmaxf(v.z, 0.f); v.w = fmaxf(v.w, 0.f);
    ((float4*)dst)[i] = v;
  }
}

// ---------------------------------------------------------------------------
extern "C" void kernel_launch(void* const* d_in, const int* in_sizes, int n_in,
                              void* d_out, int out_size, void* d_ws,
                              size_t ws_size, hipStream_t stream) {
  const float* pt        = (const float*)d_in[0];
  const float* theta     = (const float*)d_in[1];
  const float* alpha_raw = (const float*)d_in[2];
  // d_in[3] = k (==8) baked into KK=9

  unsigned short* Abuf = (unsigned short*)d_ws;     // 32 MiB bf16 graphs
  unsigned short* Wbuf =
      (unsigned short*)((char*)d_ws + (size_t)BN_TOTAL * P * P * 2);

  prep_wp_kernel<<<(D * D + 255) / 256, 256, 0, stream>>>(theta, alpha_raw, Wbuf);
  build_A_kernel<<<BN_TOTAL, 256, 0, stream>>>(pt, Abuf);
  ode_rk4_kernel<<<BN_TOTAL, 256, 0, stream>>>(pt, Abuf, Wbuf, alpha_raw,
                                               (float*)d_out);
}
